// TTTWrapper_48962627174510
// MI455X (gfx1250) — matrix operand order repaired and checked by reading, not compiled, not run on hardware
//
#include <hip/hip_runtime.h>
#include <cstdint>

// Problem constants (from reference)
#define B_   2
#define L_   4096
#define D_   2048
#define H_   32
#define KMB  16        // minibatch length K
#define HD_  64
#define NMB_ 256

typedef __attribute__((ext_vector_type(2)))  float  v2f;
typedef __attribute__((ext_vector_type(8)))  float  v8f;
typedef __attribute__((ext_vector_type(16))) __bf16 v16bf;
typedef __attribute__((ext_vector_type(4)))  unsigned int u32x4;
typedef __attribute__((ext_vector_type(8)))  int i32x8;
typedef __attribute__((ext_vector_type(4)))  int i32x4;

union BFrag { v16bf v; unsigned int u[8]; };

#if __has_builtin(__builtin_amdgcn_tensor_load_to_lds)
#define HAVE_TDM 1
#else
#define HAVE_TDM 0
#endif

__device__ __forceinline__ unsigned short f32_to_bf16_bits(float f) {
  unsigned int u = __float_as_uint(f);
  u += 0x7fffu + ((u >> 16) & 1u);           // round-to-nearest-even
  return (unsigned short)(u >> 16);
}

__device__ __forceinline__ v8f wmma_bf16(v16bf a, v16bf b, v8f c) {
  return __builtin_amdgcn_wmma_f32_16x16x32_bf16(false, a, false, b, (short)0, c, false, false);
}
__device__ __forceinline__ v8f wmma_f32(v2f a, v2f b, v8f c) {
  return __builtin_amdgcn_wmma_f32_16x16x4_f32(false, a, false, b, (short)0, c, false, false);
}

__device__ __forceinline__ float gsum8(float v) {   // sum across 8 consecutive lanes
  v += __shfl_xor(v, 1, 8);
  v += __shfl_xor(v, 2, 8);
  v += __shfl_xor(v, 4, 8);
  return v;
}

__device__ __forceinline__ unsigned lds_off32(const void* p) {
  // LDS aperture: flat shared address truncates to addr[31:0] == LDS byte offset
  return (unsigned)(size_t)p;
}

#if HAVE_TDM
// 2-D TDM load: tile (tile_d1 rows x tile_d0 elems) from a row-major tensor
// (tensor_d1 x tensor_d0, row stride stride_d0 elems) into contiguous LDS.
// D# layout per CDNA5 ISA ch.8 (higher-dim groups zero => 2-D tile).
// This toolchain exposes the 6-arg builtin: (g0, g1, g2, g3, g4, cpol).
__device__ __forceinline__ void tdm_load_2d(unsigned lds_byte_off, const void* gptr,
                                            unsigned data_size_code, // 1=2B, 2=4B
                                            unsigned tile_d0, unsigned tile_d1,
                                            unsigned tensor_d0, unsigned tensor_d1,
                                            unsigned stride_d0) {
  unsigned long long ga = (unsigned long long)(size_t)gptr;
  u32x4 g0;
  g0[0] = 1u;                                            // count=1 (valid user D#)
  g0[1] = lds_byte_off;                                  // lds_addr
  g0[2] = (unsigned)ga;                                  // global_addr[31:0]
  g0[3] = ((unsigned)(ga >> 32) & 0x01ffffffu) | 0x80000000u; // addr[56:32] | type=2
  i32x8 g1;
  g1[0] = (int)(data_size_code << 16);                   // wg_mask=0 | data_size
  g1[1] = (int)((tensor_d0 & 0xffffu) << 16);            // tensor_dim0[15:0] @bits63:48
  g1[2] = (int)(((tensor_d0 >> 16) & 0xffffu) | ((tensor_d1 & 0xffffu) << 16));
  g1[3] = (int)(((tensor_d1 >> 16) & 0xffffu) | ((tile_d0 & 0xffffu) << 16));
  g1[4] = (int)(tile_d1 & 0xffffu);                      // tile_dim1 | tile_dim2=0
  g1[5] = (int)stride_d0;                                // tensor_dim0_stride[31:0]
  g1[6] = 0;                                             // stride hi | dim1_stride lo
  g1[7] = 0;
  i32x4 z4 = {0, 0, 0, 0};
  i32x8 z8 = {0, 0, 0, 0, 0, 0, 0, 0};
  __builtin_amdgcn_tensor_load_to_lds(g0, g1, z4, z4, z8, 0);
}
#endif

// ---------------------------------------------------------------------------
// f32 -> bf16 pack (grid-stride)
// ---------------------------------------------------------------------------
__global__ void cvt_f32_bf16_kernel(const float* __restrict__ in,
                                    unsigned short* __restrict__ out, int n) {
  int i = blockIdx.x * blockDim.x + threadIdx.x;
  int stride = gridDim.x * blockDim.x;
  for (; i < n; i += stride) out[i] = f32_to_bf16_bits(in[i]);
}

// ---------------------------------------------------------------------------
// bf16 GEMM with f32 accumulate:  C[M,N] = A[M,Kd] * W[Kd,N] + bias[N]
// Block tile 128x128, 8 waves, wave tile 32x64 (2x4 WMMA accumulators).
// A tile staged by the Tensor Data Mover; B tile transposed through VGPRs.
// ---------------------------------------------------------------------------
__global__ __launch_bounds__(256)
void gemm_bf16_wmma_kernel(const unsigned short* __restrict__ A,
                           const unsigned short* __restrict__ W,
                           const float* __restrict__ bias,
                           float* __restrict__ C,
                           int M, int N, int Kd) {
  __shared__ __align__(16) unsigned short As[128 * 32];   // row-major, pitch 32
  __shared__ __align__(16) unsigned short Bs[128 * 32];   // transposed: Bs[n][k]

  const int bm = blockIdx.y * 128;
  const int bn = blockIdx.x * 128;
  const int t = threadIdx.x;
  const int wave = t >> 5, lane = t & 31;
  const int wr = wave >> 1, wc = wave & 1;        // 4x2 wave grid
  const int wm0 = wr * 32, wn0 = wc * 64;

  v8f acc[2][4];
  #pragma unroll
  for (int mi = 0; mi < 2; ++mi)
    #pragma unroll
    for (int ni = 0; ni < 4; ++ni) acc[mi][ni] = {};

  for (int k0 = 0; k0 < Kd; k0 += 32) {
#if HAVE_TDM
    // --- A tile 128x32 bf16 via TDM (wave 0 issues one 2-D descriptor) ---
    if (wave == 0)
      tdm_load_2d(lds_off32(As), A + (size_t)bm * Kd + k0,
                  /*data_size=*/1, /*tile_d0=*/32, /*tile_d1=*/128,
                  (unsigned)Kd, (unsigned)M, (unsigned)Kd);
#else
    #pragma unroll
    for (int i = 0; i < 2; ++i) {
      int c = t + i * 256;
      int r = c >> 2, cc = (c & 3) * 8;
      const uint4 ld = *(const uint4*)&A[(size_t)(bm + r) * Kd + k0 + cc];
      *(uint4*)&As[r * 32 + cc] = ld;
    }
#endif
    // --- stage W tile 32x128, transposed into Bs[n][k] ---
    #pragma unroll
    for (int i = 0; i < 2; ++i) {
      int c = t + i * 256;
      int kr = c >> 4, nc = (c & 15) * 8;
      const uint4 w4 = *(const uint4*)&W[(size_t)(k0 + kr) * N + bn + nc];
      unsigned int wd[4] = {w4.x, w4.y, w4.z, w4.w};
      #pragma unroll
      for (int j = 0; j < 4; ++j) {
        Bs[(nc + 2 * j)     * 32 + kr] = (unsigned short)(wd[j] & 0xffffu);
        Bs[(nc + 2 * j + 1) * 32 + kr] = (unsigned short)(wd[j] >> 16);
      }
      if (k0 + 32 < Kd)   // prefetch next W k-tile -> global_prefetch_b8
        __builtin_prefetch(&W[(size_t)(k0 + 32 + kr) * N + bn + nc], 0, 1);
    }
#if HAVE_TDM
    if (wave == 0) __builtin_amdgcn_s_wait_tensorcnt(0);
#endif
    __syncthreads();

    // --- build fragments: contiguous k = 16*(lane/16) + s in both A and B ---
    BFrag afr[2], bfr[4];
    const int kb  = (lane >> 4) * 16;   // A: lanes16-31 hold K=16..31
    const int kb2 = (lane >> 4) * 16;   // B: lanes16-31 hold K=16..31
    #pragma unroll
    for (int mi = 0; mi < 2; ++mi) {
      int m = wm0 + 16 * mi + (lane & 15);
      #pragma unroll
      for (int i = 0; i < 4; ++i) {
        afr[mi].u[i]     = *(const unsigned int*)&As[m * 32 + kb + 2 * i];
        afr[mi].u[4 + i] = *(const unsigned int*)&As[m * 32 + kb + 8 + 2 * i];
      }
    }
    #pragma unroll
    for (int ni = 0; ni < 4; ++ni) {
      int n = wn0 + 16 * ni + (lane & 15);
      #pragma unroll
      for (int i = 0; i < 8; ++i)
        bfr[ni].u[i] = *(const unsigned int*)&Bs[n * 32 + kb2 + 2 * i];
    }
    #pragma unroll
    for (int mi = 0; mi < 2; ++mi)
      #pragma unroll
      for (int ni = 0; ni < 4; ++ni)
        acc[mi][ni] = wmma_bf16(afr[mi].v, bfr[ni].v, acc[mi][ni]);
    __syncthreads();
  }

  // epilogue: C layout -> vgpr r: lanes0-15 M=r, lanes16-31 M=8+r; N=lane&15
  const int rowoff = (lane >> 4) * 8;
  #pragma unroll
  for (int mi = 0; mi < 2; ++mi)
    #pragma unroll
    for (int ni = 0; ni < 4; ++ni) {
      int col = bn + wn0 + 16 * ni + (lane & 15);
      float bv = bias[col];
      #pragma unroll
      for (int r = 0; r < 8; ++r) {
        int row = bm + wm0 + 16 * mi + rowoff + r;
        C[(size_t)row * N + col] = acc[mi][ni][r] + bv;
      }
    }
}

// ---------------------------------------------------------------------------
// Per-token preprocessing: L2-normalize XQ/XK, XV <- LN(XV-XKn)*w+b+XKn,
// eta[b,h,l] = sigmoid(x . lr_w[h] + lr_b[h]) / (HD*K).  One block per token.
// ---------------------------------------------------------------------------
__global__ __launch_bounds__(256)
void preprocess_kernel(const float* __restrict__ hs,
                       float* __restrict__ XQ, float* __restrict__ XK,
                       float* __restrict__ XV,
                       const float* __restrict__ lnw_g, const float* __restrict__ lnb_g,
                       const float* __restrict__ lr_w, const float* __restrict__ lr_b,
                       float* __restrict__ eta) {
  __shared__ float xrow[D_];
  const int tok = blockIdx.x;            // b*L + l
  const int b = tok / L_, l = tok - b * L_;
  const int t = threadIdx.x;
  #pragma unroll
  for (int i = 0; i < 8; ++i) xrow[t + 256 * i] = hs[(size_t)tok * D_ + t + 256 * i];
  __syncthreads();

  const int h = t >> 3, g = t & 7;       // 8 lanes per head (same wave)
  float s = 0.f;
  for (int c = g; c < D_; c += 8) s += lr_w[h * D_ + c] * xrow[c];
  s = gsum8(s);
  if (g == 0)
    eta[((size_t)b * H_ + h) * L_ + l] =
        1.f / (1.f + expf(-(s + lr_b[h]))) * (1.0f / (HD_ * KMB));

  const size_t base = (size_t)tok * D_ + h * HD_;
  float q[8], k[8], v[8];
  float sq = 0.f, sk = 0.f;
  #pragma unroll
  for (int j = 0; j < 8; ++j) {
    int c = g * 8 + j;
    q[j] = XQ[base + c]; k[j] = XK[base + c]; v[j] = XV[base + c];
    sq += q[j] * q[j];  sk += k[j] * k[j];
  }
  sq = gsum8(sq); sk = gsum8(sk);
  float invq = 1.f / fmaxf(sqrtf(sq), 1e-12f);
  float invk = 1.f / fmaxf(sqrtf(sk), 1e-12f);

  float d[8], sd = 0.f, sdd = 0.f;
  #pragma unroll
  for (int j = 0; j < 8; ++j) {
    float kn = k[j] * invk;
    d[j] = v[j] - kn;
    sd += d[j]; sdd += d[j] * d[j];
  }
  sd = gsum8(sd); sdd = gsum8(sdd);
  float mu = sd / HD_;
  float var = (sdd - HD_ * mu * mu) / (HD_ - 1);     // ddof=1
  float istd = 1.f / (sqrtf(var) + 1e-8f);

  #pragma unroll
  for (int j = 0; j < 8; ++j) {
    int c = g * 8 + j;
    float kn = k[j] * invk;
    XQ[base + c] = q[j] * invq;
    XK[base + c] = kn;
    XV[base + c] = lnw_g[h * HD_ + c] * ((d[j] - mu) * istd) + lnb_g[h * HD_ + c] + kn;
  }
}

// ---------------------------------------------------------------------------
// Sequential TTT scan.  One block (4 waves / 128 threads) per (b,h).
// All state in LDS; small matmuls via V_WMMA_F32_16X16X4_F32.
// Minibatch tiles fetched by TDM; output overwrites XV in (B,L,D) layout.
// ---------------------------------------------------------------------------
#define W1P 65   // padded pitch for W1 (column reads)

__global__ __launch_bounds__(128)
void ttt_scan_kernel(const float* __restrict__ XQ, const float* __restrict__ XK,
                     float* __restrict__ XV /* in: value target, out: y */,
                     const float* __restrict__ eta_g,
                     const float* __restrict__ W1g, const float* __restrict__ b1g,
                     const float* __restrict__ lnw_g, const float* __restrict__ lnb_g) {
  __shared__ float W1s[HD_ * W1P];
  __shared__ float b1s[HD_], lnw[HD_], lnb[HD_], etas[KMB];
  __shared__ __align__(16) float xqs[KMB * HD_];
  __shared__ __align__(16) float xks[KMB * HD_];
  __shared__ __align__(16) float xvs[KMB * HD_];
  __shared__ float z1s[KMB * HD_], z1b[KMB * HD_], grads[KMB * HD_];
  __shared__ float attnp[4][KMB * KMB];
  __shared__ float attnE[KMB * 17];
  __shared__ float b1bar[KMB * HD_];

  const int bh = blockIdx.x;
  const int b = bh / H_, h = bh - b * H_;
  const int t = threadIdx.x, wave = t >> 5, lane = t & 31;
  const int rowoff = (lane >> 4) * 8;     // C-layout row offset
  const int kh = (lane >> 4) * 2;         // f32 A/B fragment K offset

  for (int i = t; i < HD_ * HD_; i += 128)
    W1s[(i >> 6) * W1P + (i & 63)] = W1g[(size_t)h * HD_ * HD_ + i];
  if (t < HD_) {
    b1s[t] = b1g[h * HD_ + t];
    lnw[t] = lnw_g[h * HD_ + t];
    lnb[t] = lnb_g[h * HD_ + t];
  }
  __syncthreads();

  for (int n = 0; n < NMB_; ++n) {
    // ---- fetch minibatch tiles (16 x 64 f32, row stride D) ----
    const size_t rowbase = ((size_t)b * L_ + n * KMB) * D_ + h * HD_;
#if HAVE_TDM
    if (wave == 0) {
      tdm_load_2d(lds_off32(xqs), XQ + rowbase, /*data_size=*/2,
                  HD_, KMB, (unsigned)D_, (unsigned)(B_ * L_), (unsigned)D_);
      tdm_load_2d(lds_off32(xks), XK + rowbase, 2,
                  HD_, KMB, (unsigned)D_, (unsigned)(B_ * L_), (unsigned)D_);
      tdm_load_2d(lds_off32(xvs), XV + rowbase, 2,
                  HD_, KMB, (unsigned)D_, (unsigned)(B_ * L_), (unsigned)D_);
    }
#else
    for (int i = t; i < KMB * HD_; i += 128) {
      int r = i >> 6, c = i & 63;
      xqs[i] = XQ[rowbase + (size_t)r * D_ + c];
      xks[i] = XK[rowbase + (size_t)r * D_ + c];
      xvs[i] = XV[rowbase + (size_t)r * D_ + c];
    }
#endif
    if (t < KMB) etas[t] = eta_g[(size_t)bh * L_ + n * KMB + t];
#if HAVE_TDM
    if (wave == 0) __builtin_amdgcn_s_wait_tensorcnt(0);
#endif
    __syncthreads();

    // ---- Z1 = xk @ W1 + b1 : wave w owns N-tile w ----
    {
      const int nn = wave * 16 + (lane & 15);
      const int m = lane & 15;
      v8f acc = {};
      #pragma unroll
      for (int k0 = 0; k0 < HD_; k0 += 4) {
        v2f a = { xks[m * HD_ + k0 + kh], xks[m * HD_ + k0 + kh + 1] };
        v2f bb = { W1s[(k0 + kh) * W1P + nn], W1s[(k0 + kh + 1) * W1P + nn] };
        acc = wmma_f32(a, bb, acc);
      }
      #pragma unroll
      for (int r = 0; r < 8; ++r)
        z1s[(rowoff + r) * HD_ + nn] = acc[r] + b1s[nn];
    }
    __syncthreads();

    // ---- grad = ln_fused_l2_bwd(Z1, xv - xk, lnw, lnb) : 8 lanes per row ----
    {
      const int r = t >> 3, g = t & 7;
      float z[8], s1 = 0.f, s2 = 0.f;
      #pragma unroll
      for (int j = 0; j < 8; ++j) { z[j] = z1s[r * HD_ + g * 8 + j]; s1 += z[j]; s2 += z[j] * z[j]; }
      s1 = gsum8(s1); s2 = gsum8(s2);
      float mu = s1 / HD_, var = s2 / HD_ - mu * mu;
      float inv = 1.f / sqrtf(var + 1e-6f);
      float xh[8], gx[8], g1 = 0.f, g2 = 0.f;
      #pragma unroll
      for (int j = 0; j < 8; ++j) {
        int c = g * 8 + j;
        xh[j] = (z[j] - mu) * inv;
        float target = xvs[r * HD_ + c] - xks[r * HD_ + c];
        gx[j] = (lnw[c] * xh[j] + lnb[c] - target) * lnw[c];
        g1 += gx[j]; g2 += xh[j] * gx[j];
      }
      g1 = gsum8(g1); g2 = gsum8(g2);
      #pragma unroll
      for (int j = 0; j < 8; ++j)
        grads[r * HD_ + g * 8 + j] = (HD_ * gx[j] - g1 - xh[j] * g2) * (inv / HD_);
    }

    // ---- Attn1 partials: k-split across 4 waves; B = xk^T ----
    {
      const int k0 = wave * 16;
      const int m = lane & 15, nn = lane & 15;
      v8f acc = {};
      #pragma unroll
      for (int kk = 0; kk < 16; kk += 4) {
        v2f a = { xqs[m * HD_ + k0 + kk + kh], xqs[m * HD_ + k0 + kk + kh + 1] };
        v2f bb = { xks[nn * HD_ + k0 + kk + kh], xks[nn * HD_ + k0 + kk + kh + 1] };
        acc = wmma_f32(a, bb, acc);
      }
      #pragma unroll
      for (int r = 0; r < 8; ++r)
        attnp[wave][(rowoff + r) * KMB + nn] = acc[r];
    }
    __syncthreads();

    // ---- combine Attn partials, apply tril mask and -eta[j] scale ----
    for (int i = t; i < KMB * KMB; i += 128) {
      int ii = i >> 4, jj = i & 15;
      float v = attnp[0][i] + attnp[1][i] + attnp[2][i] + attnp[3][i];
      attnE[ii * 17 + jj] = (jj <= ii) ? -(etas[jj] * v) : 0.f;
    }
    // ---- b1_bar cumulative (threads 0..63, one column each) ----
    if (t < HD_) {
      float acc = b1s[t];
      #pragma unroll
      for (int i = 0; i < KMB; ++i) {
        acc -= etas[i] * grads[i * HD_ + t];
        b1bar[i * HD_ + t] = acc;
      }
    }
    __syncthreads();

    // ---- Z1_bar = xq@W1 + (-eta*Attn)@grad + b1_bar ----
    {
      const int nn = wave * 16 + (lane & 15);
      const int m = lane & 15;
      v8f acc = {};
      #pragma unroll
      for (int k0 = 0; k0 < HD_; k0 += 4) {
        v2f a = { xqs[m * HD_ + k0 + kh], xqs[m * HD_ + k0 + kh + 1] };
        v2f bb = { W1s[(k0 + kh) * W1P + nn], W1s[(k0 + kh + 1) * W1P + nn] };
        acc = wmma_f32(a, bb, acc);
      }
      #pragma unroll
      for (int k0 = 0; k0 < KMB; k0 += 4) {
        int k1 = k0 + kh;
        v2f a = { attnE[m * 17 + k1], attnE[m * 17 + k1 + 1] };
        v2f bb = { grads[k1 * HD_ + nn], grads[(k1 + 1) * HD_ + nn] };
        acc = wmma_f32(a, bb, acc);
      }
      #pragma unroll
      for (int r = 0; r < 8; ++r)
        z1b[(rowoff + r) * HD_ + nn] = acc[r] + b1bar[(rowoff + r) * HD_ + nn];
    }
    __syncthreads();

    // ---- out = xq + ln_fwd(Z1_bar) -> global y (reuses XV buffer) ----
    {
      const int r = t >> 3, g = t & 7;
      float z[8], s1 = 0.f, s2 = 0.f;
      #pragma unroll
      for (int j = 0; j < 8; ++j) { z[j] = z1b[r * HD_ + g * 8 + j]; s1 += z[j]; s2 += z[j] * z[j]; }
      s1 = gsum8(s1); s2 = gsum8(s2);
      float mu = s1 / HD_, var = s2 / HD_ - mu * mu;
      float inv = 1.f / sqrtf(var + 1e-6f);
      #pragma unroll
      for (int j = 0; j < 8; ++j) {
        int c = g * 8 + j;
        float o = xqs[r * HD_ + c] + (lnw[c] * ((z[j] - mu) * inv) + lnb[c]);
        XV[rowbase + (size_t)r * D_ + c] = o;
      }
    }

    // ---- W1 <- W1 - (eta*xk)^T @ grad : wave w owns M-tile w ----
    {
      const int m0 = wave * 16;
      const int mm = m0 + (lane & 15);
      #pragma unroll
      for (int ni = 0; ni < 4; ++ni) {
        const int nn = ni * 16 + (lane & 15);
        v8f acc;
        #pragma unroll
        for (int r = 0; r < 8; ++r)
          acc[r] = W1s[(m0 + rowoff + r) * W1P + nn];
        #pragma unroll
        for (int k0 = 0; k0 < KMB; k0 += 4) {
          int k1 = k0 + kh;
          v2f a = { -etas[k1] * xks[k1 * HD_ + mm],
                    -etas[k1 + 1] * xks[(k1 + 1) * HD_ + mm] };
          v2f bb = { grads[k1 * HD_ + nn], grads[(k1 + 1) * HD_ + nn] };
          acc = wmma_f32(a, bb, acc);
        }
        #pragma unroll
        for (int r = 0; r < 8; ++r)
          W1s[(m0 + rowoff + r) * W1P + nn] = acc[r];
      }
    }
    // ---- b1 <- b1_bar[last row] ----
    if (t < HD_) b1s[t] = b1bar[(KMB - 1) * HD_ + t];
    __syncthreads();
  }
}

// ---------------------------------------------------------------------------
// Final LayerNorm over D + pack to bf16 for the output GEMM
// ---------------------------------------------------------------------------
__global__ __launch_bounds__(256)
void postnorm_bf16_kernel(const float* __restrict__ y,
                          const float* __restrict__ pw, const float* __restrict__ pb,
                          unsigned short* __restrict__ out) {
  __shared__ float r1[256], r2[256];
  const int tok = blockIdx.x, t = threadIdx.x;
  const float* row = y + (size_t)tok * D_;
  float v[8], s1 = 0.f, s2 = 0.f;
  #pragma unroll
  for (int i = 0; i < 8; ++i) { float x = row[t + 256 * i]; v[i] = x; s1 += x; s2 += x * x; }
  r1[t] = s1; r2[t] = s2; __syncthreads();
  for (int o = 128; o > 0; o >>= 1) {
    if (t < o) { r1[t] += r1[t + o]; r2[t] += r2[t + o]; }
    __syncthreads();
  }
  float mu = r1[0] / D_, var = r2[0] / D_ - mu * mu;
  float inv = rsqrtf(var + 1e-6f);
  #pragma unroll
  for (int i = 0; i < 8; ++i) {
    int c = t + 256 * i;
    float o = (v[i] - mu) * inv * pw[c] + pb[c];
    out[(size_t)tok * D_ + c] = f32_to_bf16_bits(o);
  }
}

// ---------------------------------------------------------------------------
extern "C" void kernel_launch(void* const* d_in, const int* in_sizes, int n_in,
                              void* d_out, int out_size, void* d_ws, size_t ws_size,
                              hipStream_t stream) {
  (void)in_sizes; (void)n_in; (void)out_size; (void)ws_size;
  const float* hs    = (const float*)d_in[0];
  const float* wq_w  = (const float*)d_in[1];
  const float* wq_b  = (const float*)d_in[2];
  const float* wk_w  = (const float*)d_in[3];
  const float* wk_b  = (const float*)d_in[4];
  const float* wv_w  = (const float*)d_in[5];
  const float* wv_b  = (const float*)d_in[6];
  const float* wo_w  = (const float*)d_in[7];
  const float* wo_b  = (const float*)d_in[8];
  const float* lnw   = (const float*)d_in[9];    // ttt_norm_weight (H,HD)
  const float* lnb   = (const float*)d_in[10];   // ttt_norm_bias   (H,HD)
  const float* lr_w  = (const float*)d_in[11];   // (H,1,D)
  const float* lr_b  = (const float*)d_in[12];   // (H,1)
  const float* W1g   = (const float*)d_in[13];   // (H,HD,HD)
  const float* b1g   = (const float*)d_in[14];   // (H,1,HD)
  const float* pnw   = (const float*)d_in[15];
  const float* pnb   = (const float*)d_in[16];

  const size_t ML = (size_t)B_ * L_;             // 8192 rows
  char* ws = (char*)d_ws;
  unsigned short* hs_bf = (unsigned short*)ws;                          // ML*D bf16 (reused as y_norm bf16)
  unsigned short* w_bf  = (unsigned short*)(ws + ML * D_ * 2);          // 4 * D*D bf16
  unsigned short* wqbf = w_bf;
  unsigned short* wkbf = w_bf + (size_t)D_ * D_;
  unsigned short* wvbf = w_bf + 2 * (size_t)D_ * D_;
  unsigned short* wobf = w_bf + 3 * (size_t)D_ * D_;
  float* XQ  = (float*)(ws + ML * D_ * 2 + 4 * (size_t)D_ * D_ * 2);
  float* XK  = XQ + ML * D_;
  float* XV  = XK + ML * D_;                     // becomes y after scan
  float* eta = XV + ML * D_;                     // (B,H,L)

  // 1) pack to bf16
  cvt_f32_bf16_kernel<<<8192, 256, 0, stream>>>(hs,   hs_bf, (int)(ML * D_));
  cvt_f32_bf16_kernel<<<2048, 256, 0, stream>>>(wq_w, wqbf,  D_ * D_);
  cvt_f32_bf16_kernel<<<2048, 256, 0, stream>>>(wk_w, wkbf,  D_ * D_);
  cvt_f32_bf16_kernel<<<2048, 256, 0, stream>>>(wv_w, wvbf,  D_ * D_);
  cvt_f32_bf16_kernel<<<2048, 256, 0, stream>>>(wo_w, wobf,  D_ * D_);

  // 2) QKV projection GEMMs (bf16 WMMA, f32 accumulate, TDM-staged A tiles)
  dim3 gg(D_ / 128, (unsigned)(ML / 128));
  gemm_bf16_wmma_kernel<<<gg, 256, 0, stream>>>(hs_bf, wqbf, wq_b, XQ, (int)ML, D_, D_);
  gemm_bf16_wmma_kernel<<<gg, 256, 0, stream>>>(hs_bf, wkbf, wk_b, XK, (int)ML, D_, D_);
  gemm_bf16_wmma_kernel<<<gg, 256, 0, stream>>>(hs_bf, wvbf, wv_b, XV, (int)ML, D_, D_);

  // 3) per-token preprocessing + learning rates
  preprocess_kernel<<<(unsigned)ML, 256, 0, stream>>>(hs, XQ, XK, XV, lnw, lnb, lr_w, lr_b, eta);

  // 4) sequential TTT scan (f32 WMMA + TDM tile fetch), output in-place over XV
  ttt_scan_kernel<<<B_ * H_, 128, 0, stream>>>(XQ, XK, XV, eta, W1g, b1g, lnw, lnb);

  // 5) post LayerNorm + pack bf16 (reuse hs_bf), then output projection
  postnorm_bf16_kernel<<<(unsigned)ML, 256, 0, stream>>>(XV, pnw, pnb, hs_bf);
  gemm_bf16_wmma_kernel<<<gg, 256, 0, stream>>>(hs_bf, wobf, wo_b, (float*)d_out, (int)ML, D_, D_);
}
